// ResidualHeteroGATConv_50620484551193
// MI455X (gfx1250) — matrix operand order repaired and checked by reading, not compiled, not run on hardware
//
#include <hip/hip_runtime.h>
#include <hip/hip_bf16.h>
#include <stdint.h>

// ---------------------------------------------------------------------------
// Types for CDNA5 WMMA
// ---------------------------------------------------------------------------
typedef __attribute__((ext_vector_type(16))) __bf16 v16bf;
typedef __attribute__((ext_vector_type(8)))  float  v8f;

#define NH 100000
#define NF 100000
#define INF_ 256
#define HH 4
#define DD 128
#define HD 512
#define E1N 200000
#define E2N 200000

// ---------------------------------------------------------------------------
// Small utility kernels
// ---------------------------------------------------------------------------
__global__ void k_cvt_bf16(const float* __restrict__ x, __bf16* __restrict__ y, long n) {
  long i = (long)blockIdx.x * blockDim.x + threadIdx.x;
  if (i < n) y[i] = (__bf16)x[i];
}

__global__ void k_zero_u32(unsigned* __restrict__ p, long n) {
  long i = (long)blockIdx.x * blockDim.x + threadIdx.x;
  if (i < n) p[i] = 0u;
}

// rst[i] = bias[i mod 512]  (bias pre-added so zero-edge nodes match reference)
__global__ void k_bias_init(float* __restrict__ p, const float* __restrict__ b, long n) {
  long i = (long)blockIdx.x * blockDim.x + threadIdx.x;
  if (i < n) p[i] = b[i & 511];
}

// Pack f32 W[K=256, N] row-major into WMMA-B-native bf16 chunks:
// Bpk[((nt*8+kk)*32+lane)*16 + c] = W[(kk*32 + (c>>3)*16 + (lane>>4)*8 + (c&7))*N + nt*16 + (lane&15)]
__global__ void k_pack_b(const float* __restrict__ W, __bf16* __restrict__ Bpk, int N) {
  long total = (long)256 * N;
  long i = (long)blockIdx.x * blockDim.x + threadIdx.x;
  if (i >= total) return;
  int c    = (int)(i & 15);
  int lane = (int)((i >> 4) & 31);
  int kk   = (int)((i >> 9) & 7);
  int nt   = (int)(i >> 12);
  int n = nt * 16 + (lane & 15);
  int k = kk * 32 + (c >> 3) * 16 + (lane >> 4) * 8 + (c & 7);
  Bpk[i] = (__bf16)W[(long)k * N + n];
}

// Wl[k][h] = sum_d W[k, h*128+d] * avec[h*128+d]   (collapses dst-projection GEMM)
__global__ void k_make_attn_w(const float* __restrict__ W, const float* __restrict__ avec,
                              float* __restrict__ out) {
  int i = blockIdx.x * blockDim.x + threadIdx.x;
  if (i >= 256 * 4) return;
  int k = i >> 2, h = i & 3;
  float acc = 0.f;
  const float* wr = W + (long)k * 512 + h * 128;
  const float* av = avec + h * 128;
#pragma unroll 4
  for (int d = 0; d < 128; ++d) acc += wr[d] * av[d];
  out[k * 4 + h] = acc;
}

// el[n,h] = feat[n,:] @ Wl[:,h]   (Wl is [256][4] row-major)
__global__ void k_attn_scores(const float* __restrict__ feat, const float* __restrict__ Wl,
                              float* __restrict__ out, int N) {
  int n = blockIdx.x * blockDim.x + threadIdx.x;
  if (n >= N) return;
  const float* f = feat + (long)n * INF_;
  const float4* wl = (const float4*)Wl;
  float4 acc = make_float4(0.f, 0.f, 0.f, 0.f);
#pragma unroll 4
  for (int k = 0; k < INF_; ++k) {
    float x = f[k];
    float4 w = wl[k];
    acc.x += x * w.x; acc.y += x * w.y; acc.z += x * w.z; acc.w += x * w.w;
  }
  *(float4*)(out + (long)n * 4) = acc;
}

// ---------------------------------------------------------------------------
// WMMA GEMM: C[M,N] f32 = A[M,256] bf16 @ Bpk (packed bf16), K = 256 fixed.
// block = 256 threads (8 waves), each wave computes a 16x64 tile.
// grid.x = ceil(M/128), grid.y = N/64.
// ---------------------------------------------------------------------------
__global__ void k_gemm_wmma(const __bf16* __restrict__ A, const __bf16* __restrict__ Bpk,
                            float* __restrict__ C, int M, int N) {
  const int lane = threadIdx.x & 31;
  const int wave = threadIdx.x >> 5;
  const int row0 = (blockIdx.x * 8 + wave) * 16;
  if (row0 >= M) return;                 // wave-uniform: EXEC stays all-ones
  const int col0 = blockIdx.y * 64;
  const int nt0  = blockIdx.y * 4;

  const int m  = lane & 15;
  const int kh = lane >> 4;
  const __bf16* Arow = A + (long)(row0 + m) * 256;

  v8f acc[4];
#pragma unroll
  for (int t = 0; t < 4; ++t) acc[t] = (v8f){0.f,0.f,0.f,0.f,0.f,0.f,0.f,0.f};

#pragma unroll
  for (int kk = 0; kk < 8; ++kk) {
    const int k0 = kk * 32;
    union { v16bf v; uint4 q[2]; } a;
    a.q[0] = *(const uint4*)(Arow + k0 + kh * 8);
    a.q[1] = *(const uint4*)(Arow + k0 + 16 + kh * 8);
#pragma unroll
    for (int t = 0; t < 4; ++t) {
      union { v16bf v; uint4 q[2]; } b;
      const uint4* bp = (const uint4*)(Bpk + ((((long)(nt0 + t) * 8 + kk) * 32 + lane) << 4));
      b.q[0] = bp[0];
      b.q[1] = bp[1];
      acc[t] = __builtin_amdgcn_wmma_f32_16x16x32_bf16(
          false, a.v, false, b.v, (short)0, acc[t], false, false);
    }
  }

  const int l15 = lane & 15;
  const int hi8 = (lane >> 4) * 8;
#pragma unroll
  for (int t = 0; t < 4; ++t) {
    float* crow = C + (long)(row0 + hi8) * N + col0 + t * 16 + l15;
#pragma unroll
    for (int r = 0; r < 8; ++r) crow[(long)r * N] = acc[t][r];
  }
}

// ---------------------------------------------------------------------------
// Edge softmax (order-preserving float<->u32 key for atomic max)
// ---------------------------------------------------------------------------
__device__ __forceinline__ unsigned fkey(float f) {
  unsigned u = __float_as_uint(f);
  return (u >> 31) ? ~u : (u | 0x80000000u);
}
__device__ __forceinline__ float fdekey(unsigned k) {
  unsigned u = (k >> 31) ? (k & 0x7FFFFFFFu) : ~k;
  return __uint_as_float(u);
}

__global__ void k_edge_logits_max(const int* __restrict__ src, const int* __restrict__ dst,
                                  const float* __restrict__ el, const float* __restrict__ er,
                                  float* __restrict__ elog, unsigned* __restrict__ mkey, int E) {
  int e = blockIdx.x * blockDim.x + threadIdx.x;
  if (e >= E) return;
  int s = src[e], d = dst[e];
#pragma unroll
  for (int h = 0; h < 4; ++h) {
    float v = el[(long)s * 4 + h] + er[(long)d * 4 + h];
    v = v > 0.f ? v : 0.2f * v;            // LeakyReLU(0.2)
    elog[(long)e * 4 + h] = v;
    atomicMax(&mkey[(long)d * 4 + h], fkey(v));
  }
}

__global__ void k_edge_exp_sum(const int* __restrict__ dst, const float* __restrict__ elog,
                               const unsigned* __restrict__ mkey, float* __restrict__ ee,
                               float* __restrict__ ssum, int E) {
  int e = blockIdx.x * blockDim.x + threadIdx.x;
  if (e >= E) return;
  int d = dst[e];
#pragma unroll
  for (int h = 0; h < 4; ++h) {
    float mm = fdekey(mkey[(long)d * 4 + h]);
    float x = __expf(elog[(long)e * 4 + h] - mm);
    ee[(long)e * 4 + h] = x;
    unsafeAtomicAdd(&ssum[(long)d * 4 + h], x);
  }
}

// 128 threads per edge, 2 edges per 256-thread block: thread covers (h, 4 floats of d)
__global__ void k_edge_scatter(const int* __restrict__ src, const int* __restrict__ dst,
                               const float* __restrict__ ee, const float* __restrict__ ssum,
                               const float* __restrict__ fs, float* __restrict__ rst) {
  int t = threadIdx.x;
  int e = blockIdx.x * 2 + (t >> 7);
  int tt = t & 127;
  int h = tt >> 5;
  int d0 = (tt & 31) * 4;
  int s = src[e], d = dst[e];
  float a = ee[(long)e * 4 + h] / ssum[(long)d * 4 + h];
  const float4 f = *(const float4*)(fs + (long)s * HD + h * DD + d0);
  float* r = rst + (long)d * HD + h * DD + d0;
  unsafeAtomicAdd(r + 0, a * f.x);
  unsafeAtomicAdd(r + 1, a * f.y);
  unsafeAtomicAdd(r + 2, a * f.z);
  unsafeAtomicAdd(r + 3, a * f.w);
}

// ---------------------------------------------------------------------------
// Tail: head-mean + residual(proj staged in io) + LayerNorm + exact GELU
// block = 128 (one thread per d), grid = num nodes
// ---------------------------------------------------------------------------
__global__ void k_tail(const float* __restrict__ rst, float* __restrict__ io,
                       const float* __restrict__ pb, const float* __restrict__ g,
                       const float* __restrict__ be) {
  const int n = blockIdx.x;
  const int d = threadIdx.x;
  const float* r = rst + (long)n * HD;
  float c = 0.25f * (r[d] + r[128 + d] + r[256 + d] + r[384 + d])
          + io[(long)n * DD + d] + pb[d];

  __shared__ float sh[128];
  sh[d] = c;
  __syncthreads();
#pragma unroll
  for (int st = 64; st > 0; st >>= 1) {
    if (d < st) sh[d] += sh[d + st];
    __syncthreads();
  }
  float mu = sh[0] * (1.f / 128.f);
  __syncthreads();
  float dc = c - mu;
  sh[d] = dc * dc;
  __syncthreads();
#pragma unroll
  for (int st = 64; st > 0; st >>= 1) {
    if (d < st) sh[d] += sh[d + st];
    __syncthreads();
  }
  float var = sh[0] * (1.f / 128.f);
  float y = dc * rsqrtf(var + 1e-5f) * g[d] + be[d];
  // exact GELU
  io[(long)n * DD + d] = 0.5f * y * (1.f + erff(y * 0.70710678118654752f));
}

// ---------------------------------------------------------------------------
// Launch
// ---------------------------------------------------------------------------
extern "C" void kernel_launch(void* const* d_in, const int* in_sizes, int n_in,
                              void* d_out, int out_size, void* d_ws, size_t ws_size,
                              hipStream_t stream) {
  const float* feat_host = (const float*)d_in[0];
  const float* feat_flow = (const float*)d_in[1];
  const float* W1  = (const float*)d_in[2];
  const float* al1 = (const float*)d_in[3];
  const float* ar1 = (const float*)d_in[4];
  const float* b1  = (const float*)d_in[5];
  const float* W2  = (const float*)d_in[6];
  const float* al2 = (const float*)d_in[7];
  const float* ar2 = (const float*)d_in[8];
  const float* b2  = (const float*)d_in[9];
  const float* pWh = (const float*)d_in[10];
  const float* pbh = (const float*)d_in[11];
  const float* pWf = (const float*)d_in[12];
  const float* pbf = (const float*)d_in[13];
  const float* g_host  = (const float*)d_in[14];
  const float* be_host = (const float*)d_in[15];
  const float* g_flow  = (const float*)d_in[16];
  const float* be_flow = (const float*)d_in[17];
  const int* e1_src = (const int*)d_in[18];
  const int* e1_dst = (const int*)d_in[19];
  const int* e2_src = (const int*)d_in[20];
  const int* e2_dst = (const int*)d_in[21];

  float* out_host = (float*)d_out;                       // [NH,128]
  float* out_flow = (float*)d_out + (long)NH * DD;       // [NF,128]

  // ---- workspace layout ----
  char* w = (char*)d_ws;
  size_t off = 0;
  auto alloc = [&](size_t bytes) -> char* {
    char* p = w + off;
    off = (off + bytes + 255) & ~(size_t)255;
    return p;
  };
  __bf16* featHb = (__bf16*)alloc((size_t)NH * INF_ * 2);
  __bf16* featFb = (__bf16*)alloc((size_t)NF * INF_ * 2);
  __bf16* wpk1   = (__bf16*)alloc((size_t)256 * HD * 2);
  __bf16* wpk2   = (__bf16*)alloc((size_t)256 * HD * 2);
  __bf16* ppkH   = (__bf16*)alloc((size_t)256 * DD * 2);
  __bf16* ppkF   = (__bf16*)alloc((size_t)256 * DD * 2);
  float* Wl1 = (float*)alloc(256 * 4 * 4);
  float* Wr1 = (float*)alloc(256 * 4 * 4);
  float* Wl2 = (float*)alloc(256 * 4 * 4);
  float* Wr2 = (float*)alloc(256 * 4 * 4);
  float* fs1 = (float*)alloc((size_t)NH * HD * 4);
  float* fs2 = (float*)alloc((size_t)NF * HD * 4);
  float* el1 = (float*)alloc((size_t)NH * 4 * 4);
  float* er1 = (float*)alloc((size_t)NF * 4 * 4);
  float* el2 = (float*)alloc((size_t)NF * 4 * 4);
  float* er2 = (float*)alloc((size_t)NH * 4 * 4);
  float* elog1 = (float*)alloc((size_t)E1N * 4 * 4);
  float* ee1   = (float*)alloc((size_t)E1N * 4 * 4);
  float* elog2 = (float*)alloc((size_t)E2N * 4 * 4);
  float* ee2   = (float*)alloc((size_t)E2N * 4 * 4);
  unsigned* m1 = (unsigned*)alloc((size_t)NF * 4 * 4);
  float*    s1 = (float*)alloc((size_t)NF * 4 * 4);
  unsigned* m2 = (unsigned*)alloc((size_t)NH * 4 * 4);
  float*    s2 = (float*)alloc((size_t)NH * 4 * 4);
  float* rst1 = (float*)alloc((size_t)NF * HD * 4);  // flow aggregation (rel h2f)
  float* rst2 = (float*)alloc((size_t)NH * HD * 4);  // host aggregation (rel f2h)
  (void)ws_size; (void)in_sizes; (void)n_in; (void)out_size;

  const int TB = 256;
  auto blk = [](long n, int t) { return (unsigned)((n + t - 1) / t); };

  // 1) fp32 -> bf16 feats
  k_cvt_bf16<<<blk((long)NH * INF_, TB), TB, 0, stream>>>(feat_host, featHb, (long)NH * INF_);
  k_cvt_bf16<<<blk((long)NF * INF_, TB), TB, 0, stream>>>(feat_flow, featFb, (long)NF * INF_);

  // 2) pack weights into WMMA-B-native layout
  k_pack_b<<<blk((long)256 * HD, TB), TB, 0, stream>>>(W1, wpk1, HD);
  k_pack_b<<<blk((long)256 * HD, TB), TB, 0, stream>>>(W2, wpk2, HD);
  k_pack_b<<<blk((long)256 * DD, TB), TB, 0, stream>>>(pWh, ppkH, DD);
  k_pack_b<<<blk((long)256 * DD, TB), TB, 0, stream>>>(pWf, ppkF, DD);

  // 3) collapse attention vectors: Wl = W*al, Wr = W*ar  ([256,4] each)
  k_make_attn_w<<<4, 256, 0, stream>>>(W1, al1, Wl1);
  k_make_attn_w<<<4, 256, 0, stream>>>(W1, ar1, Wr1);
  k_make_attn_w<<<4, 256, 0, stream>>>(W2, al2, Wl2);
  k_make_attn_w<<<4, 256, 0, stream>>>(W2, ar2, Wr2);

  // 4) WMMA GEMMs: message features + residual projections (proj staged in d_out)
  {
    dim3 g1(blk(NH, 128), HD / 64);
    k_gemm_wmma<<<g1, 256, 0, stream>>>(featHb, wpk1, fs1, NH, HD);
    dim3 g2(blk(NF, 128), HD / 64);
    k_gemm_wmma<<<g2, 256, 0, stream>>>(featFb, wpk2, fs2, NF, HD);
    dim3 g3(blk(NH, 128), DD / 64);
    k_gemm_wmma<<<g3, 256, 0, stream>>>(featHb, ppkH, out_host, NH, DD);
    dim3 g4(blk(NF, 128), DD / 64);
    k_gemm_wmma<<<g4, 256, 0, stream>>>(featFb, ppkF, out_flow, NF, DD);
  }

  // 5) attention scores el/er (fp32, tiny GEMV against [256,4])
  k_attn_scores<<<blk(NH, TB), TB, 0, stream>>>(feat_host, Wl1, el1, NH);
  k_attn_scores<<<blk(NF, TB), TB, 0, stream>>>(feat_flow, Wr1, er1, NF);
  k_attn_scores<<<blk(NF, TB), TB, 0, stream>>>(feat_flow, Wl2, el2, NF);
  k_attn_scores<<<blk(NH, TB), TB, 0, stream>>>(feat_host, Wr2, er2, NH);

  // 6) reset accumulators every call (graph replays!)
  k_zero_u32<<<blk((long)NF * 4, TB), TB, 0, stream>>>(m1, (long)NF * 4);
  k_zero_u32<<<blk((long)NF * 4, TB), TB, 0, stream>>>((unsigned*)s1, (long)NF * 4);
  k_zero_u32<<<blk((long)NH * 4, TB), TB, 0, stream>>>(m2, (long)NH * 4);
  k_zero_u32<<<blk((long)NH * 4, TB), TB, 0, stream>>>((unsigned*)s2, (long)NH * 4);
  k_bias_init<<<blk((long)NF * HD, TB), TB, 0, stream>>>(rst1, b1, (long)NF * HD);
  k_bias_init<<<blk((long)NH * HD, TB), TB, 0, stream>>>(rst2, b2, (long)NH * HD);

  // 7) edge softmax, 3 passes per relation
  k_edge_logits_max<<<blk(E1N, TB), TB, 0, stream>>>(e1_src, e1_dst, el1, er1, elog1, m1, E1N);
  k_edge_logits_max<<<blk(E2N, TB), TB, 0, stream>>>(e2_src, e2_dst, el2, er2, elog2, m2, E2N);
  k_edge_exp_sum<<<blk(E1N, TB), TB, 0, stream>>>(e1_dst, elog1, m1, ee1, s1, E1N);
  k_edge_exp_sum<<<blk(E2N, TB), TB, 0, stream>>>(e2_dst, elog2, m2, ee2, s2, E2N);
  k_edge_scatter<<<E1N / 2, 256, 0, stream>>>(e1_src, e1_dst, ee1, s1, fs1, rst1);
  k_edge_scatter<<<E2N / 2, 256, 0, stream>>>(e2_src, e2_dst, ee2, s2, fs2, rst2);

  // 8) tails: host uses rst2 (f2h aggregation), flow uses rst1 (h2f aggregation)
  k_tail<<<NH, DD, 0, stream>>>(rst2, out_host, pbh, g_host, be_host);
  k_tail<<<NF, DD, 0, stream>>>(rst1, out_flow, pbf, g_flow, be_flow);
}
